// KalmanNetNN_77249281785929
// MI455X (gfx1250) — matrix operand order repaired and checked by reading, compiled, not run on hardware
//
#include <hip/hip_runtime.h>
#include <math.h>

#define BROWS 8192

typedef unsigned short u16;
typedef __attribute__((ext_vector_type(16))) __bf16 v16bf;
typedef __attribute__((ext_vector_type(16))) u16    us16;
typedef __attribute__((ext_vector_type(8)))  u16    us8;
typedef __attribute__((ext_vector_type(8)))  float  v8f;

__device__ __forceinline__ u16 f2bf(float f) {
    unsigned u = __builtin_bit_cast(unsigned, f);
    u += 0x7fffu + ((u >> 16) & 1u);            // round-to-nearest-even
    return (u16)(u >> 16);
}
__device__ __forceinline__ float sigm(float x) { return 1.0f / (1.0f + expf(-x)); }

__device__ __forceinline__ v8f wmma_bf16(const us16& a, const us16& b, v8f c) {
    return __builtin_amdgcn_wmma_f32_16x16x32_bf16(
        false, __builtin_bit_cast(v16bf, a),
        false, __builtin_bit_cast(v16bf, b),
        (short)0, c, false, false);
}

// Assemble A fragment (16x32 bf16, contiguous k = 16*half + s) from two contiguous 8-element runs
__device__ __forceinline__ us16 loadA(const u16* rowPtr, int k0, int half) {
    const us8* pa = (const us8*)(rowPtr + k0 + (half << 4));
    us8 lo = pa[0];          // K = k0+half*16 .. +7
    us8 hi = pa[1];          // K = k0+half*16+8 .. +7
    us16 a;
#pragma unroll
    for (int j = 0; j < 8; ++j) { a[j] = lo[j]; a[j + 8] = hi[j]; }
    return a;
}

// ---------------------------------------------------------------------------
// Y[B,N] = act(X[B,KP]bf16 @ Wb[N,KP]bf16^T + bias)   all shapes compile-time
// X and Wb are zero-padded in K, so the k-loop is completely guard-free.
// wave-per-16x16 tile; grid=(512, ceil(nTiles/4)); block=128
// ---------------------------------------------------------------------------
template <int N, int KP, int LDXP, int RELU, int OUTBF16>
__global__ __launch_bounds__(128) void gemm_bf16_kernel(
    const u16* __restrict__ X, const u16* __restrict__ Wb,
    const float* __restrict__ bias, void* __restrict__ Yv, int ldy)
{
    const int lane  = threadIdx.x & 31;
    const int wave  = threadIdx.x >> 5;
    const int nTile = blockIdx.y * 4 + wave;
    constexpr int nTiles = (N + 15) >> 4;
    if (nTile >= nTiles) return;                 // wave-uniform

    const int half  = lane >> 4;
    const int l15   = lane & 15;
    const int mBase = blockIdx.x << 4;
    const int nRow  = nTile * 16 + l15;
    const bool nValid = (nRow < N);
    const int  nR   = nValid ? nRow : (N - 1);   // clamp: junk cols never stored

    const u16* Xr = X  + (long)(mBase + l15) * LDXP;
    const u16* Wr = Wb + (long)nR * KP;

    v8f acc = {};
#pragma unroll
    for (int k0 = 0; k0 < KP; k0 += 32) {
        us16 au = loadA(Xr, k0, half);
        us16 bu = *(const us16*)(Wr + k0 + (half << 4));  // 32B aligned
        acc = wmma_bf16(au, bu, acc);
    }

    const float bv = bias[nR];
#pragma unroll
    for (int r = 0; r < 8; ++r) {                // D: m=r+8*half, n=l15
        int   m = mBase + r + (half << 3);
        float v = acc[r] + bv;
        if (RELU) v = fmaxf(v, 0.0f);
        if (nValid) {
            if (OUTBF16) ((u16*)Yv)[(long)m * ldy + nRow] = f2bf(v);
            else         ((float*)Yv)[(long)m * ldy + nRow] = v;
        }
    }
}

// ---------------------------------------------------------------------------
// Fused FC2: hid = relu(X[B,181]@W1^T + b1); acc += hid @ W2^T
// X: [B,192] bf16 padded; W1b: [7240,192] bf16; W2b: [90,7264] bf16 padded.
// wave = (mTile, hiddenGroup of 4); partials via f32 global atomics.
// ---------------------------------------------------------------------------
__global__ __launch_bounds__(128) void fc2_fused_kernel(
    const u16* __restrict__ X, const u16* __restrict__ W1b,
    const float* __restrict__ b1, const u16* __restrict__ W2b,
    float* __restrict__ acc)
{
    __shared__ u16 hid[4][16][32];               // per-wave bf16 bounce tile

    const int lane  = threadIdx.x & 31;
    const int wave  = threadIdx.x >> 5;
    const int t     = blockIdx.x * 4 + wave;     // 0..2047
    const int mTile = t >> 2;
    const int group = t & 3;
    const int mBase = mTile << 4;
    const int half  = lane >> 4;
    const int l15   = lane & 15;

    constexpr int K1P = 192, NH = 7240, KP2 = 7264, N2 = 90;

    // Preload all 6 A1 fragments (X zero-padded -> no guards)
    const u16* Xr = X + (long)(mBase + l15) * K1P;
    us16 a1[6];
#pragma unroll
    for (int ks = 0; ks < 6; ++ks) a1[ks] = loadA(Xr, ks * 32, half);

    v8f c2[6];
    v8f vz = {};
#pragma unroll
    for (int i = 0; i < 6; ++i) c2[i] = vz;

    constexpr int CH_TOT = (NH + 31) / 32;       // 227
    constexpr int CH_PER = (CH_TOT + 3) / 4;     // 57
    int ch0 = group * CH_PER;
    int chE = ch0 + CH_PER; if (chE > CH_TOT) chE = CH_TOT;

    for (int ch = ch0; ch < chE; ++ch) {
        const int hBase = ch * 32;

        // stage 1: two 16-wide hidden tiles
#pragma unroll
        for (int hh = 0; hh < 2; ++hh) {
            const int  n1  = hBase + hh * 16 + l15;
            const bool nv  = (n1 < NH);
            const int  n1c = nv ? n1 : (NH - 1);
            const u16* Wr  = W1b + (long)n1c * K1P;
            v8f c1 = {};
#pragma unroll
            for (int ks = 0; ks < 6; ++ks) {
                us16 bu = *(const us16*)(Wr + ks * 32 + (half << 4));
                c1 = wmma_bf16(a1[ks], bu, c1);
            }
            const float bb   = b1[n1c];
            const int   hcol = hh * 16 + l15;
#pragma unroll
            for (int r = 0; r < 8; ++r) {        // force 0 for padded hidden cols
                float v = fmaxf(c1[r] + bb, 0.0f);
                hid[wave][r + (half << 3)][hcol] = nv ? f2bf(v) : (u16)0;
            }
        }
        __asm__ volatile("s_wait_dscnt 0" ::: "memory");

        // stage 2: A2 from LDS (A layout runs), 6 N-tiles of W2
        us16 a2;
        {
            const us8* pl = (const us8*)&hid[wave][l15][half << 4];
            us8 lo = pl[0], hi = pl[1];
#pragma unroll
            for (int j = 0; j < 8; ++j) { a2[j] = lo[j]; a2[j + 8] = hi[j]; }
        }
#pragma unroll
        for (int t6 = 0; t6 < 6; ++t6) {
            const int n2  = t6 * 16 + l15;
            const int n2c = (n2 < N2) ? n2 : (N2 - 1);
            const u16* Wr2 = W2b + (long)n2c * KP2;
            us16 bu = *(const us16*)(Wr2 + hBase + (half << 4));
            c2[t6] = wmma_bf16(a2, bu, c2[t6]);
        }
    }

#pragma unroll
    for (int t6 = 0; t6 < 6; ++t6) {
        int n2 = t6 * 16 + l15;
        if (n2 < N2) {
#pragma unroll
            for (int r = 0; r < 8; ++r) {
                int m = mBase + r + (half << 3);
                atomicAdd(&acc[(long)m * N2 + n2], c2[t6][r]);
            }
        }
    }
}

// ---------------------------------------------------------------------------
// Helper kernels
// ---------------------------------------------------------------------------
__global__ void convw_kernel(const float* __restrict__ src, u16* __restrict__ dst,
                             int N, int K, int KP)
{
    long i = (long)blockIdx.x * blockDim.x + threadIdx.x;
    if (i >= (long)N * KP) return;
    int n = (int)(i / KP), k = (int)(i - (long)n * KP);
    dst[i] = (k < K) ? f2bf(src[(long)n * K + k]) : (u16)0;
}

__global__ void zero_u32_kernel(unsigned* __restrict__ p, long n) {
    long i = (long)blockIdx.x * blockDim.x + threadIdx.x;
    if (i < n) p[i] = 0u;
}

__device__ __forceinline__ void l2emit(const float* in, int n, u16* o) {
    float s = 0.0f;
    for (int i = 0; i < n; ++i) s += in[i] * in[i];
    float inv = 1.0f / fmaxf(sqrtf(s), 1e-12f);
    for (int i = 0; i < 32; ++i) o[i] = (i < n) ? f2bf(in[i] * inv) : (u16)0;
}

__global__ void norm_kernel(const float* __restrict__ od, const float* __restrict__ oid,
                            const float* __restrict__ fe, const float* __restrict__ fu,
                            u16* __restrict__ xn)
{
    int b = blockIdx.x * blockDim.x + threadIdx.x;
    if (b >= BROWS) return;
    u16* o = xn + (long)b * 128;
    l2emit(od  + (long)b * 9,  9,  o + 0);    // obs_diff
    l2emit(oid + (long)b * 9,  9,  o + 32);   // obs_innov_diff
    l2emit(fe  + (long)b * 10, 10, o + 64);   // fw_evol_diff
    l2emit(fu  + (long)b * 10, 10, o + 96);   // fw_update_diff
}

// gh = h0 @ Whh^T + bhh  (batch-constant: h0 is eye or zero)
__global__ void gh_kernel(const float* __restrict__ QWhh, const float* __restrict__ Qbhh,
                          const float* __restrict__ Sigbhh,
                          const float* __restrict__ RWhh, const float* __restrict__ Rbhh,
                          const float* __restrict__ SWhh, const float* __restrict__ Sbhh,
                          float* __restrict__ gh)
{
    int i = blockIdx.x * blockDim.x + threadIdx.x;
    if (i < 300) {
        float s = Qbhh[i];
        for (int d = 0; d < 10; ++d) s += QWhh[(long)i * 100 + 11 * d];
        gh[i] = s;
    } else if (i < 600) {
        gh[i] = Sigbhh[i - 300];
    } else if (i < 843) {
        int j = i - 600; float s = Rbhh[j];
        for (int d = 0; d < 9; ++d) s += RWhh[(long)j * 81 + 10 * d];
        gh[i] = s;
    } else if (i < 1086) {
        int j = i - 843; float s = Sbhh[j];
        for (int d = 0; d < 9; ++d) s += SWhh[(long)j * 81 + 10 * d];
        gh[i] = s;
    }
}

// GRU combine: one f32 dest (d_out) + up to three bf16 dests (padded buffers)
__global__ void gru_combine_kernel(const float* __restrict__ gi, int H,
                                   const float* __restrict__ ghc, int idPeriod,
                                   float* df, int ldf,
                                   u16* d0, int ld0, u16* d1, int ld1, u16* d2, int ld2)
{
    long idx = (long)blockIdx.x * blockDim.x + threadIdx.x;
    if (idx >= (long)BROWS * H) return;
    int b = (int)(idx / H), j = (int)(idx - (long)b * H);
    const float* g = gi + (long)b * 3 * H;
    float r = sigm(g[j]          + ghc[j]);
    float z = sigm(g[H + j]      + ghc[H + j]);
    float n = tanhf(g[2 * H + j] + r * ghc[2 * H + j]);
    float h0 = (idPeriod > 0 && (j % idPeriod) == 0) ? 1.0f : 0.0f;
    float h  = (1.0f - z) * n + z * h0;
    u16 hb = f2bf(h);
    if (df) df[(long)b * ldf + j] = h;
    if (d0) d0[(long)b * ld0 + j] = hb;
    if (d1) d1[(long)b * ld1 + j] = hb;
    if (d2) d2[(long)b * ld2 + j] = hb;
}

__global__ void fc2_epilogue_kernel(const float* __restrict__ acc,
                                    const float* __restrict__ b2,
                                    float* __restrict__ kg,       // d_out, ld 90
                                    u16* __restrict__ fc3in)      // [B,192] bf16, +81
{
    long idx = (long)blockIdx.x * blockDim.x + threadIdx.x;
    if (idx >= (long)BROWS * 90) return;
    int b = (int)(idx / 90), n = (int)(idx - (long)b * 90);
    float v = acc[idx] + b2[n];
    kg[(long)b * 90 + n] = v;
    fc3in[(long)b * 192 + 81 + n] = f2bf(v);
}

// ---------------------------------------------------------------------------
extern "C" void kernel_launch(void* const* d_in, const int* in_sizes, int n_in,
                              void* d_out, int out_size, void* d_ws, size_t ws_size,
                              hipStream_t stream)
{
    (void)in_sizes; (void)n_in; (void)out_size; (void)ws_size;
    const long B = BROWS;
    const float* obs   = (const float*)d_in[0];
    const float* innov = (const float*)d_in[1];
    const float* evol  = (const float*)d_in[2];
    const float* upd   = (const float*)d_in[3];
    const float* QWih  = (const float*)d_in[4];
    const float* QWhh  = (const float*)d_in[5];
    const float* Qbih  = (const float*)d_in[6];
    const float* Qbhh  = (const float*)d_in[7];
    const float* GWih  = (const float*)d_in[8];
    const float* Gbih  = (const float*)d_in[10];
    const float* Gbhh  = (const float*)d_in[11];
    const float* SWih  = (const float*)d_in[12];
    const float* SWhh  = (const float*)d_in[13];
    const float* Sbih  = (const float*)d_in[14];
    const float* Sbhh  = (const float*)d_in[15];
    const float* RWih  = (const float*)d_in[16];
    const float* RWhh  = (const float*)d_in[17];
    const float* Rbih  = (const float*)d_in[18];
    const float* Rbhh  = (const float*)d_in[19];
    const float* fc1W  = (const float*)d_in[20];
    const float* fc1b  = (const float*)d_in[21];
    const float* fc2W1 = (const float*)d_in[22];
    const float* fc2b1 = (const float*)d_in[23];
    const float* fc2W2 = (const float*)d_in[24];
    const float* fc2b2 = (const float*)d_in[25];
    const float* fc3W  = (const float*)d_in[26];
    const float* fc3b  = (const float*)d_in[27];
    const float* fc4W  = (const float*)d_in[28];
    const float* fc4b  = (const float*)d_in[29];
    const float* fc5W  = (const float*)d_in[30];
    const float* fc5b  = (const float*)d_in[31];
    const float* fc6W  = (const float*)d_in[32];
    const float* fc6b  = (const float*)d_in[33];
    const float* fc7W  = (const float*)d_in[34];
    const float* fc7b  = (const float*)d_in[35];
    const float* fc8W  = (const float*)d_in[36];
    const float* fc8b  = (const float*)d_in[37];

    float* out = (float*)d_out;
    float* oKG   = out;                 // [B,90]
    float* ohS   = out + B * 90;        // [B,81]
    float* ohSig = out + B * 171;       // [B,100] h_Sigma_new
    float* ohQ   = out + B * 271;       // [B,100]
    float* ohR   = out + B * 371;       // [B,81]

    // ---- workspace layout (bf16 region first, then f32 region) ----
    u16* p = (u16*)d_ws;
    u16* f5    = p; p += B * 64;        // fc5_out     K=50  ->64
    u16* f8    = p; p += B * 64;        // fc8_out     K=45  ->64
    u16* sigin = p; p += B * 160;       // [h_Q|fc6]   K=150 ->160
    u16* hsig  = p; p += B * 128;       // h_Sigma     K=100 ->128
    u16* fc2in = p; p += B * 192;       // [hSig|h_S]  K=181 ->192
    u16* fc4in = p; p += B * 224;       // [hSig|fc3]  K=200 ->224
    u16* sinb  = p; p += B * 224;       // [fc1|fc7|hR]K=207 ->224
    u16* fc3in = p; p += B * 192;       // [h_S|FC2]   K=171 ->192
    const long zeroU16 = B * 1248;      // everything above must be pre-zeroed
    u16* xn    = p; p += B * 128;       // 4 normalized segs, fully written
    u16* wq  = p; p += 300L * 64;
    u16* wsg = p; p += 300L * 160;
    u16* wss = p; p += 243L * 224;
    u16* wr  = p; p += 243L * 64;
    u16* wf1c= p; p += 81L * 128;       // fc1
    u16* w3  = p; p += 100L * 192;
    u16* w4  = p; p += 100L * 224;
    u16* w5  = p; p += 50L * 32;
    u16* w6  = p; p += 50L * 32;
    u16* w7  = p; p += 45L * 32;
    u16* w8  = p; p += 45L * 32;
    u16* wb1 = p; p += 7240L * 192;     // fc2_W1
    u16* wb2 = p; p += 90L * 7264;      // fc2_W2
    float* gi   = (float*)p;
    float* facc = gi + B * 300;
    float* gh   = facc + B * 90;

    const dim3 blk(128);
    auto ggrid = [](int N) { int nt = (N + 15) >> 4; return dim3(512, (unsigned)((nt + 3) >> 2)); };
    const int CB = 256;
    auto eb = [&](long n) { return dim3((unsigned)((n + CB - 1) / CB)); };
    auto conv = [&](const float* s, u16* d, int N, int K, int KP) {
        convw_kernel<<<eb((long)N * KP), dim3(CB), 0, stream>>>(s, d, N, K, KP);
    };

    // zero padded activation buffers + fc2 accumulator (poisoned by harness)
    zero_u32_kernel<<<eb(zeroU16 / 2), dim3(CB), 0, stream>>>((unsigned*)f5, zeroU16 / 2);
    zero_u32_kernel<<<eb(B * 90), dim3(CB), 0, stream>>>((unsigned*)facc, B * 90);

    // weight bf16 conversion (once per call; zero-padded K)
    conv(fc5W, w5, 50, 10, 32);    conv(fc6W, w6, 50, 10, 32);
    conv(fc7W, w7, 45, 9, 32);     conv(fc8W, w8, 45, 9, 32);
    conv(QWih, wq, 300, 50, 64);   conv(GWih, wsg, 300, 150, 160);
    conv(fc1W, wf1c, 81, 100, 128);
    conv(RWih, wr, 243, 45, 64);   conv(SWih, wss, 243, 207, 224);
    conv(fc3W, w3, 100, 171, 192); conv(fc4W, w4, 100, 200, 224);
    conv(fc2W1, wb1, 7240, 181, 192);
    conv(fc2W2, wb2, 90, 7240, 7264);

    gh_kernel<<<dim3(5), dim3(256), 0, stream>>>(QWhh, Qbhh, Gbhh, RWhh, Rbhh, SWhh, Sbhh, gh);
    norm_kernel<<<eb(B), dim3(CB), 0, stream>>>(obs, innov, evol, upd, xn);

    // FC5(upd) -> Q_gi -> h_Q ; FC6(evol)
    gemm_bf16_kernel<50, 32, 128, 1, 1><<<ggrid(50), blk, 0, stream>>>(xn + 96, w5, fc5b, f5, 64);
    gemm_bf16_kernel<300, 64, 64, 0, 0><<<ggrid(300), blk, 0, stream>>>(f5, wq, Qbih, gi, 300);
    gemm_bf16_kernel<50, 32, 128, 1, 1><<<ggrid(50), blk, 0, stream>>>(xn + 64, w6, fc6b, sigin + 100, 160);
    gru_combine_kernel<<<eb(B * 100), dim3(CB), 0, stream>>>(gi, 100, gh, 11,
        ohQ, 100, sigin, 160, nullptr, 0, nullptr, 0);

    // Sigma_gi -> h_Sigma
    gemm_bf16_kernel<300, 160, 160, 0, 0><<<ggrid(300), blk, 0, stream>>>(sigin, wsg, Gbih, gi, 300);
    gru_combine_kernel<<<eb(B * 100), dim3(CB), 0, stream>>>(gi, 100, gh + 300, 0,
        nullptr, 0, hsig, 128, fc2in, 192, fc4in, 224);

    // FC7(obs), FC8(innov), FC1(h_Sigma)
    gemm_bf16_kernel<45, 32, 128, 1, 1><<<ggrid(45), blk, 0, stream>>>(xn + 0, w7, fc7b, sinb + 81, 224);
    gemm_bf16_kernel<45, 32, 128, 1, 1><<<ggrid(45), blk, 0, stream>>>(xn + 32, w8, fc8b, f8, 64);
    gemm_bf16_kernel<81, 128, 128, 1, 1><<<ggrid(81), blk, 0, stream>>>(hsig, wf1c, fc1b, sinb, 224);

    // R_gi -> h_R
    gemm_bf16_kernel<243, 64, 64, 0, 0><<<ggrid(243), blk, 0, stream>>>(f8, wr, Rbih, gi, 243);
    gru_combine_kernel<<<eb(B * 81), dim3(CB), 0, stream>>>(gi, 81, gh + 600, 10,
        ohR, 81, sinb + 126, 224, nullptr, 0, nullptr, 0);

    // S_gi -> h_S
    gemm_bf16_kernel<243, 224, 224, 0, 0><<<ggrid(243), blk, 0, stream>>>(sinb, wss, Sbih, gi, 243);
    gru_combine_kernel<<<eb(B * 81), dim3(CB), 0, stream>>>(gi, 81, gh + 843, 10,
        ohS, 81, fc2in + 100, 192, fc3in, 192, nullptr, 0);

    // FC2 fused (dominant GEMM pair) + bias epilogue -> KG + fc3in[:,81:]
    fc2_fused_kernel<<<dim3(512), blk, 0, stream>>>(fc2in, wb1, fc2b1, wb2, facc);
    fc2_epilogue_kernel<<<eb(B * 90), dim3(CB), 0, stream>>>(facc, fc2b2, oKG, fc3in);

    // FC3 -> fc4in[:,100:200]; FC4 -> h_Sigma_new
    gemm_bf16_kernel<100, 192, 192, 1, 1><<<ggrid(100), blk, 0, stream>>>(fc3in, w3, fc3b, fc4in + 100, 224);
    gemm_bf16_kernel<100, 224, 224, 1, 0><<<ggrid(100), blk, 0, stream>>>(fc4in, w4, fc4b, ohSig, 100);
}
